// ImpalaCNNAgent_78958678769675
// MI455X (gfx1250) — matrix operand order repaired and checked by reading, not compiled, not run on hardware
//
#include <hip/hip_runtime.h>
#include <hip/hip_bf16.h>

// ---------------------------------------------------------------------------
// IMPALA CNN agent forward pass for MI455X (gfx1250, wave32, WMMA).
//
// Convs: implicit GEMM on v_wmma_f32_16x16x32_f16, LDS-staged pre-swizzled
// fragments (K ordered k = tap*Cpad + ci -> every B fragment is ONE
// contiguous 32B ds_load). Each wave register-tiles 2x2 (cout x pixel)
// accumulators: 4 WMMAs per 4 ds_loads, independent accs hide the
// WMMA->WMMA RAW hazard. f16 activations, f32 accumulate.
// FC / GRU-input-projection: WMMA GEMM with 2 m-tiles per wave.
// CBAM / GRU recurrence / heads: scalar f32 (tiny, latency-bound).
// ---------------------------------------------------------------------------

typedef __attribute__((ext_vector_type(16))) _Float16 v16h;
typedef __attribute__((ext_vector_type(8)))  _Float16 v8h;
typedef __attribute__((ext_vector_type(8)))  float    v8f;

#define TB 512            // T*B rows
#define NEG_BIG (-3.0e38f)

// ---------------------------------------------------------------------------
// 3x3 conv (pad 1). Block = 256 threads = 8 waves; block covers 256
// contiguous output pixels (full rows: 256/W rows) x ALL output channels.
// Wave w computes pixel tiles {2w, 2w+1} x all (COUT/16) cout tiles.
//
// K ordering: k = tap*CPAD + ci  (tap = r*3+s in [0,9), ci channel).
// LDS input tile layout  : inS[row][x][ci]  (ci contiguous)
// LDS weight layout      : wA pre-swizzled per (mtile, chunk, lane) -> v16h
// Zero padding beyond Ktot / Cin enforced on the A side only (B may read
// arbitrary in-bounds LDS; A zeros kill the products).
//
// WMMA lane layouts per CDNA5 ISA 7.12.2:
//   A f16 16x32: lane l -> M=l&15; elem e -> K = (e<8?0:16)+(l>>4)*8+(e&7)
//   B f16 32x16: lane l -> N=l&15; elem e -> K as for A (this code uses K = (l>>4)*16 + e for both A and B)
//   C/D f32    : lane l -> N=l&15; reg  r -> M = r + 8*(l>>4)
// ---------------------------------------------------------------------------
template<typename TIN, int CIN, int CPAD, int COUT, int H, int W,
         bool RELU_IN, bool HAS_RES>
__global__ __launch_bounds__(256)
void conv3x3_wmma(const TIN* __restrict__ in, const float* __restrict__ wgt,
                  const float* __restrict__ bias, const _Float16* __restrict__ res,
                  _Float16* __restrict__ out)
{
    constexpr int KTOT   = 9 * CPAD;
    constexpr int NCHUNK = (KTOT + 31) / 32;
    constexpr int MT     = COUT / 16;      // cout tiles per wave (1 or 2)
    constexpr int NROWS  = 256 / W;        // output rows covered by this block
    constexpr int RALLOC = NROWS + 3;      // staged rows + 1 guaranteed-zero row
    constexpr int WP     = W + 2;          // +-1 column halo
    constexpr int CSH    = (CPAD == 8 ? 3 : (CPAD == 16 ? 4 : 5));

    __shared__ v16h wA[MT * NCHUNK * 32];                   // pre-swizzled A
    __shared__ __align__(64) _Float16 inS[RALLOC * WP * CPAD];

    const int tid = threadIdx.x;
    const int img = blockIdx.z;
    const int pb  = blockIdx.x * 256;      // first output pixel of block
    const int y0  = pb / W;                // first output row of block

    // ---- stage weights into WMMA A layout (zero beyond Ktot / Cin) -------
    {
        _Float16* wAh = (_Float16*)wA;
        for (int idx = tid; idx < MT * NCHUNK * 512; idx += 256) {
            int e  = idx & 15;
            int l  = (idx >> 4) & 31;
            int c  = (idx >> 9) % NCHUNK;
            int mt = idx / (NCHUNK * 512);
            int k  = c * 32 + (l >> 4) * 16 + e;
            int tap = k >> CSH;
            int ci  = k & (CPAD - 1);
            float v = 0.0f;
            if (tap < 9 && ci < CIN) {
                int r = tap / 3, s = tap - 3 * (tap / 3);
                v = wgt[(((mt * 16 + (l & 15)) * CIN + ci) * 3 + r) * 3 + s];
            }
            wAh[idx] = (_Float16)v;
        }
    }

    // ---- stage haloed input tile: rows y0-1 .. y0+NROWS (+1 zero row) ----
    for (int idx = tid; idx < RALLOC * CPAD * WP; idx += 256) {
        int x  = idx % WP;
        int t  = idx / WP;
        int ci = t & (CPAD - 1);
        int rr = t >> CSH;
        int gy = y0 - 1 + rr;
        int gx = x - 1;
        float v = 0.0f;
        if (rr < NROWS + 2 && ci < CIN && gy >= 0 && gy < H && gx >= 0 && gx < W) {
            v = (float)in[((size_t)(img * CIN + ci) * H + gy) * W + gx];
            if (RELU_IN) v = v > 0.0f ? v : 0.0f;
        }
        inS[(rr * WP + x) * CPAD + ci] = (_Float16)v;
    }
    __syncthreads();

    // ---- compute: 8 waves x (MT x 2) register tiles ----------------------
    const int wv   = tid >> 5;
    const int lane = tid & 31;
    const int half = lane >> 4;
    const int l15  = lane & 15;

    int  ylocT[2], x0T[2];
#pragma unroll
    for (int pt = 0; pt < 2; ++pt) {
        int p0 = pb + (2 * wv + pt) * 16;
        ylocT[pt] = p0 / W - y0;           // 0..NROWS-1
        x0T[pt]   = p0 % W;
    }

    v8f acc[MT][2];
#pragma unroll
    for (int mt = 0; mt < MT; ++mt)
#pragma unroll
        for (int pt = 0; pt < 2; ++pt) acc[mt][pt] = (v8f){};

#pragma unroll
    for (int c = 0; c < NCHUNK; ++c) {
        v16h a[MT];
#pragma unroll
        for (int mt = 0; mt < MT; ++mt) a[mt] = wA[(mt * NCHUNK + c) * 32 + lane];

        v16h b[2];
#pragma unroll
        for (int pt = 0; pt < 2; ++pt) {
            const int xw = x0T[pt] + l15;
            if constexpr (CPAD == 32) {
                // tap == c (compile-time), 16 elems = channels half*16..+15
                b[pt] = *(const v16h*)
                    &inS[((ylocT[pt] + (c / 3)) * WP + (xw + (c % 3))) * 32 + half * 16];
            } else if constexpr (CPAD == 16) {
                int tap = 2 * c + half;
                int r = (tap * 11) >> 5;   // tap/3 for tap in [0,11]
                int s = tap - 3 * r;
                b[pt] = *(const v16h*)&inS[((ylocT[pt] + r) * WP + (xw + s)) * 16];
            } else {                        // CPAD == 8: two taps per fragment
                int t0 = 4 * c + 2 * half;
                int r0 = (t0 * 11) >> 5, s0 = t0 - 3 * r0;
                int t1 = t0 + 1;
                int r1 = (t1 * 11) >> 5, s1 = t1 - 3 * r1;
                v8h lo = *(const v8h*)&inS[((ylocT[pt] + r0) * WP + (xw + s0)) * 8];
                v8h hi = *(const v8h*)&inS[((ylocT[pt] + r1) * WP + (xw + s1)) * 8];
#pragma unroll
                for (int e = 0; e < 8; ++e) { b[pt][e] = lo[e]; b[pt][8 + e] = hi[e]; }
            }
        }

#pragma unroll
        for (int mt = 0; mt < MT; ++mt)
#pragma unroll
            for (int pt = 0; pt < 2; ++pt)
                acc[mt][pt] = __builtin_amdgcn_wmma_f32_16x16x32_f16(
                    false, a[mt], false, b[pt], (short)0, acc[mt][pt], false, false);
    }

    // ---- epilogue: bias (+ residual) -> f16 NCHW -------------------------
#pragma unroll
    for (int mt = 0; mt < MT; ++mt) {
#pragma unroll
        for (int pt = 0; pt < 2; ++pt) {
            int yw  = y0 + ylocT[pt];
            int xg  = x0T[pt] + l15;
#pragma unroll
            for (int r = 0; r < 8; ++r) {
                int mo = mt * 16 + r + half * 8;
                size_t idx = ((size_t)(img * COUT + mo) * H + yw) * W + xg;
                float v = acc[mt][pt][r] + bias[mo];
                if (HAS_RES) v += (float)res[idx];
                out[idx] = (_Float16)v;
            }
        }
    }
}

// ---------------------------------------------------------------------------
// 3x3 stride-2 maxpool, pad 1. f16 -> f16.
// ---------------------------------------------------------------------------
__global__ void maxpool3s2_kernel(const _Float16* __restrict__ in,
                                  _Float16* __restrict__ out,
                                  int C, int Hin, int Win, int nimg, int img_off)
{
    int Hp = Hin >> 1, Wp = Win >> 1;
    size_t total = (size_t)nimg * C * Hp * Wp;
    for (size_t i = (size_t)blockIdx.x * blockDim.x + threadIdx.x; i < total;
         i += (size_t)gridDim.x * blockDim.x) {
        int wp = (int)(i % Wp); size_t t = i / Wp;
        int hp = (int)(t % Hp); t /= Hp;
        int c  = (int)(t % C);
        int im = (int)(t / C);
        float mx = NEG_BIG;
        for (int dy = 0; dy < 3; ++dy) {
            int yy = 2 * hp - 1 + dy;
            if (yy < 0 || yy >= Hin) continue;
            for (int dx = 0; dx < 3; ++dx) {
                int xx = 2 * wp - 1 + dx;
                if (xx < 0 || xx >= Win) continue;
                float v = (float)in[((size_t)(im * C + c) * Hin + yy) * Win + xx];
                mx = fmaxf(mx, v);
            }
        }
        out[((size_t)((im + img_off) * C + c) * Hp + hp) * Wp + wp] = (_Float16)mx;
    }
}

// ---------------------------------------------------------------------------
// WMMA GEMM: out[M,N] = act(A[M,K](f16) @ B[N,K]^T(f32) + bias[N])
// One wave per (2 m-tiles) x (1 n-tile): B fragment (f32 weights, the
// expensive side) is loaded once and reused by two WMMAs.
// ---------------------------------------------------------------------------
template<bool RELU, typename TOUT>
__global__ __launch_bounds__(32)
void gemm_wmma(const _Float16* __restrict__ A, int lda,
               const float* __restrict__ B,   // N x K row-major
               const float* __restrict__ bias,
               TOUT* __restrict__ out, int ldc, int K)
{
    const int lane = threadIdx.x;
    const int half = lane >> 4;
    const int l15  = lane & 15;
    const int n0 = blockIdx.x * 16;
    const int m0 = blockIdx.y * 32;

    const int mA0 = m0 + l15;
    const int mA1 = m0 + 16 + l15;
    const int nB  = n0 + l15;

    v8f acc0 = {}, acc1 = {};
    for (int kc = 0; kc < K; kc += 32) {
        v16h a0, a1, b;
#pragma unroll
        for (int e = 0; e < 16; ++e) {
            int ka = kc + half * 16 + e;
            a0[e] = A[(size_t)mA0 * lda + ka];
            a1[e] = A[(size_t)mA1 * lda + ka];
            int kb = kc + half * 16 + e;
            b[e] = (_Float16)B[(size_t)nB * K + kb];
        }
        acc0 = __builtin_amdgcn_wmma_f32_16x16x32_f16(false, a0, false, b,
                                                      (short)0, acc0, false, false);
        acc1 = __builtin_amdgcn_wmma_f32_16x16x32_f16(false, a1, false, b,
                                                      (short)0, acc1, false, false);
    }
#pragma unroll
    for (int r = 0; r < 8; ++r) {
        int no = n0 + l15;
        float bi = bias[no];
        int mo0 = m0 + r + half * 8;
        float v0 = acc0[r] + bi;
        if (RELU) v0 = v0 > 0.0f ? v0 : 0.0f;
        out[(size_t)mo0 * ldc + no] = (TOUT)v0;
        int mo1 = m0 + 16 + r + half * 8;
        float v1 = acc1[r] + bi;
        if (RELU) v1 = v1 > 0.0f ? v1 : 0.0f;
        out[(size_t)mo1 * ldc + no] = (TOUT)v1;
    }
}

// ---------------------------------------------------------------------------
// CBAM per image: channel attention (32->8->32 MLP on mean/max, sigmoid)
// then 7x7 spatial attention. Relu of stage-2 output folded into the load.
// ---------------------------------------------------------------------------
__global__ __launch_bounds__(256)
void cbam_kernel(const _Float16* __restrict__ x,
                 const float* __restrict__ fc1,   // (8,32)
                 const float* __restrict__ fc2,   // (32,8)
                 const float* __restrict__ spw,   // (1,2,7,7)
                 _Float16* __restrict__ out)
{
    __shared__ float xr[32][256];
    __shared__ float s_mean[32], s_max[32];
    __shared__ float h1[8];
    __shared__ float ca[32];
    __shared__ float sp[2][256];
    __shared__ float sa[256];

    const int tid = threadIdx.x;
    const int img = blockIdx.x;
    const _Float16* xi = x + (size_t)img * 8192;

    for (int i = tid; i < 8192; i += 256) {
        float v = (float)xi[i];
        xr[i >> 8][i & 255] = v > 0.0f ? v : 0.0f;
    }
    __syncthreads();

    if (tid < 32) {
        float s = 0.0f, m = NEG_BIG;
        for (int p = 0; p < 256; ++p) { float v = xr[tid][p]; s += v; m = fmaxf(m, v); }
        s_mean[tid] = s * (1.0f / 256.0f);
        s_max[tid]  = m;
    }
    __syncthreads();

    if (tid < 8) {
        float am = 0.0f, ax = 0.0f;
        for (int c = 0; c < 32; ++c) {
            float w = fc1[tid * 32 + c];
            am += w * s_mean[c];
            ax += w * s_max[c];
        }
        am = am > 0.0f ? am : 0.0f;
        ax = ax > 0.0f ? ax : 0.0f;
        h1[tid] = am + ax;               // fc2 linear: fc2@h1m + fc2@h1x
    }
    __syncthreads();

    if (tid < 32) {
        float v = 0.0f;
        for (int t = 0; t < 8; ++t) v += fc2[tid * 8 + t] * h1[t];
        ca[tid] = 1.0f / (1.0f + expf(-v));
    }
    __syncthreads();

    for (int i = tid; i < 8192; i += 256) xr[i >> 8][i & 255] *= ca[i >> 8];
    __syncthreads();

    {
        int p = tid;
        float s = 0.0f, m = NEG_BIG;
        for (int c = 0; c < 32; ++c) { float v = xr[c][p]; s += v; m = fmaxf(m, v); }
        sp[0][p] = s * (1.0f / 32.0f);
        sp[1][p] = m;
    }
    __syncthreads();

    {
        int p = tid, y = p >> 4, x0 = p & 15;
        float acc = 0.0f;
        for (int chn = 0; chn < 2; ++chn)
            for (int dy = 0; dy < 7; ++dy) {
                int yy = y + dy - 3;
                if (yy < 0 || yy >= 16) continue;
                for (int dx = 0; dx < 7; ++dx) {
                    int xx = x0 + dx - 3;
                    if (xx < 0 || xx >= 16) continue;
                    acc += spw[(chn * 7 + dy) * 7 + dx] * sp[chn][yy * 16 + xx];
                }
            }
        sa[p] = 1.0f / (1.0f + expf(-acc));
    }
    __syncthreads();

    _Float16* o = out + (size_t)img * 8192;
    for (int i = tid; i < 8192; i += 256)
        o[i] = (_Float16)(xr[i >> 8][i & 255] * sa[i & 255]);
}

// ---------------------------------------------------------------------------
// GRU recurrence (GI precomputed by WMMA GEMM). 1 block, 1024 thr, 64 steps.
// ---------------------------------------------------------------------------
__global__ __launch_bounds__(1024)
void gru_kernel(const float* __restrict__ GI,    // (512,384)
                const float* __restrict__ whh,   // (384,128)
                const float* __restrict__ bhh,   // (384)
                const float* __restrict__ done,  // (512)
                const float* __restrict__ h0,    // (1,8,128)
                float* __restrict__ hs)          // (512,128)
{
    __shared__ float h[8][128];
    __shared__ float gh[8][384];

    const int tid = threadIdx.x;
    const int b = tid >> 7, j = tid & 127;
    h[b][j] = h0[b * 128 + j];
    __syncthreads();

    for (int t = 0; t < 64; ++t) {
        h[b][j] *= (1.0f - done[t * 8 + b]);
        __syncthreads();

        for (int o = tid; o < 3072; o += 1024) {
            int bb = o / 384, g = o - bb * 384;
            float s = bhh[g];
            const float* wr = whh + g * 128;
            for (int k = 0; k < 128; ++k) s += wr[k] * h[bb][k];
            gh[bb][g] = s;
        }
        __syncthreads();

        {
            const float* gi = GI + (size_t)(t * 8 + b) * 384;
            float r = 1.0f / (1.0f + expf(-(gi[j]       + gh[b][j])));
            float z = 1.0f / (1.0f + expf(-(gi[128 + j] + gh[b][128 + j])));
            float n = tanhf(gi[256 + j] + r * gh[b][256 + j]);
            float hn = (1.0f - z) * n + z * h[b][j];
            h[b][j] = hn;
            hs[(size_t)(t * 8 + b) * 128 + j] = hn;
        }
        __syncthreads();
    }
}

// ---------------------------------------------------------------------------
// Actor/critic heads.
// ---------------------------------------------------------------------------
__global__ void heads_kernel(const float* __restrict__ hs,
                             const float* __restrict__ aw, const float* __restrict__ ab,
                             const float* __restrict__ cw, const float* __restrict__ cb,
                             float* __restrict__ out)
{
    int i = blockIdx.x * blockDim.x + threadIdx.x;
    if (i >= TB * 4) return;
    int row = i >> 2, c = i & 3;
    const float* h = hs + (size_t)row * 128;
    float s;
    const float* w;
    if (c < 3) { w = aw + c * 128; s = ab[c]; }
    else       { w = cw;           s = cb[0]; }
    for (int k = 0; k < 128; ++k) s += w[k] * h[k];
    out[i] = s;
}

// ---------------------------------------------------------------------------
// Host-side launch sequence
// ---------------------------------------------------------------------------
static inline int pool_blocks(size_t total) { return (int)((total + 255) / 256); }

extern "C" void kernel_launch(void* const* d_in, const int* in_sizes, int n_in,
                              void* d_out, int out_size, void* d_ws, size_t ws_size,
                              hipStream_t stream)
{
    (void)in_sizes; (void)n_in; (void)out_size; (void)ws_size;

    const float* x         = (const float*)d_in[0];
    const float* done      = (const float*)d_in[1];
    const float* gru_state = (const float*)d_in[2];
    const float* s0_cw = (const float*)d_in[3];
    const float* s0_cb = (const float*)d_in[4];
    const float* s0_rw = (const float*)d_in[5];
    const float* s0_rb = (const float*)d_in[6];
    const float* s1_cw = (const float*)d_in[7];
    const float* s1_cb = (const float*)d_in[8];
    const float* s1_rw = (const float*)d_in[9];
    const float* s1_rb = (const float*)d_in[10];
    const float* s2_cw = (const float*)d_in[11];
    const float* s2_cb = (const float*)d_in[12];
    const float* s2_rw = (const float*)d_in[13];
    const float* s2_rb = (const float*)d_in[14];
    const float* cbam_fc1 = (const float*)d_in[15];
    const float* cbam_fc2 = (const float*)d_in[16];
    const float* cbam_spw = (const float*)d_in[17];
    const float* fc_w  = (const float*)d_in[18];
    const float* fc_b  = (const float*)d_in[19];
    const float* gru_wih = (const float*)d_in[20];
    const float* gru_whh = (const float*)d_in[21];
    const float* gru_bih = (const float*)d_in[22];
    const float* gru_bhh = (const float*)d_in[23];
    const float* actor_w  = (const float*)d_in[24];
    const float* actor_b  = (const float*)d_in[25];
    const float* critic_w = (const float*)d_in[26];
    const float* critic_b = (const float*)d_in[27];

    // ---- workspace layout (static offsets, f16 activations) ----
    char* ws = (char*)d_ws;
    const size_t MB = 1024u * 1024u;
    _Float16* convfull = (_Float16*)(ws);                   // <= 34 MB (64-img chunk)
    _Float16* actA     = (_Float16*)(ws + 36 * MB);         // <= 68 MB
    _Float16* actB     = (_Float16*)(ws + 104 * MB);        // <= 68 MB
    _Float16* feat     = (_Float16*)(ws + 172 * MB);        // 512*256 f16
    float*    GI       = (float*)   (ws + 173 * MB);        // 512*384 f32
    float*    hs       = (float*)   (ws + 174 * MB);        // 512*128 f32

    const int CHUNK = 64;            // images per conv+pool chunk
    const int NCH = TB / CHUNK;      // 8 chunks

    // ================= Stage 0: conv 6->16 @128, pool ->64, resblocks =====
    for (int ch = 0; ch < NCH; ++ch) {
        conv3x3_wmma<float, 6, 8, 16, 128, 128, false, false>
            <<<dim3(128 * 128 / 256, 1, CHUNK), 256, 0, stream>>>(
                x + (size_t)ch * CHUNK * 6 * 128 * 128,
                s0_cw, s0_cb, nullptr, convfull);
        size_t tot = (size_t)CHUNK * 16 * 64 * 64;
        maxpool3s2_kernel<<<pool_blocks(tot), 256, 0, stream>>>(
            convfull, actA, 16, 128, 128, CHUNK, ch * CHUNK);
    }
    for (int rb = 0; rb < 2; ++rb) {
        const float* w1 = s0_rw + (size_t)(2 * rb) * 16 * 16 * 9;
        const float* w2 = s0_rw + (size_t)(2 * rb + 1) * 16 * 16 * 9;
        const float* b1 = s0_rb + (2 * rb) * 16;
        const float* b2 = s0_rb + (2 * rb + 1) * 16;
        conv3x3_wmma<_Float16, 16, 16, 16, 64, 64, true, false>
            <<<dim3(64 * 64 / 256, 1, TB), 256, 0, stream>>>(
                actA, w1, b1, nullptr, actB);
        conv3x3_wmma<_Float16, 16, 16, 16, 64, 64, true, true>
            <<<dim3(64 * 64 / 256, 1, TB), 256, 0, stream>>>(
                actB, w2, b2, actA, actA);
    }

    // ================= Stage 1: conv 16->32 @64, pool ->32, resblocks =====
    for (int ch = 0; ch < NCH; ++ch) {
        conv3x3_wmma<_Float16, 16, 16, 32, 64, 64, false, false>
            <<<dim3(64 * 64 / 256, 1, CHUNK), 256, 0, stream>>>(
                actA + (size_t)ch * CHUNK * 16 * 64 * 64,
                s1_cw, s1_cb, nullptr, convfull);
        size_t tot = (size_t)CHUNK * 32 * 32 * 32;
        maxpool3s2_kernel<<<pool_blocks(tot), 256, 0, stream>>>(
            convfull, actB, 32, 64, 64, CHUNK, ch * CHUNK);
    }
    for (int rb = 0; rb < 2; ++rb) {
        const float* w1 = s1_rw + (size_t)(2 * rb) * 32 * 32 * 9;
        const float* w2 = s1_rw + (size_t)(2 * rb + 1) * 32 * 32 * 9;
        const float* b1 = s1_rb + (2 * rb) * 32;
        const float* b2 = s1_rb + (2 * rb + 1) * 32;
        conv3x3_wmma<_Float16, 32, 32, 32, 32, 32, true, false>
            <<<dim3(32 * 32 / 256, 1, TB), 256, 0, stream>>>(
                actB, w1, b1, nullptr, actA);
        conv3x3_wmma<_Float16, 32, 32, 32, 32, 32, true, true>
            <<<dim3(32 * 32 / 256, 1, TB), 256, 0, stream>>>(
                actA, w2, b2, actB, actB);
    }

    // ================= Stage 2: conv 32->32 @32, pool ->16, resblocks =====
    for (int ch = 0; ch < NCH; ++ch) {
        conv3x3_wmma<_Float16, 32, 32, 32, 32, 32, false, false>
            <<<dim3(32 * 32 / 256, 1, CHUNK), 256, 0, stream>>>(
                actB + (size_t)ch * CHUNK * 32 * 32 * 32,
                s2_cw, s2_cb, nullptr, convfull);
        size_t tot = (size_t)CHUNK * 32 * 16 * 16;
        maxpool3s2_kernel<<<pool_blocks(tot), 256, 0, stream>>>(
            convfull, actA, 32, 32, 32, CHUNK, ch * CHUNK);
    }
    for (int rb = 0; rb < 2; ++rb) {
        const float* w1 = s2_rw + (size_t)(2 * rb) * 32 * 32 * 9;
        const float* w2 = s2_rw + (size_t)(2 * rb + 1) * 32 * 32 * 9;
        const float* b1 = s2_rb + (2 * rb) * 32;
        const float* b2 = s2_rb + (2 * rb + 1) * 32;
        conv3x3_wmma<_Float16, 32, 32, 32, 16, 16, true, false>
            <<<dim3(16 * 16 / 256, 1, TB), 256, 0, stream>>>(
                actA, w1, b1, nullptr, actB);
        conv3x3_wmma<_Float16, 32, 32, 32, 16, 16, true, true>
            <<<dim3(16 * 16 / 256, 1, TB), 256, 0, stream>>>(
                actB, w2, b2, actA, actA);
    }

    // ================= CBAM (relu folded into load) =======================
    cbam_kernel<<<TB, 256, 0, stream>>>(actA, cbam_fc1, cbam_fc2, cbam_spw, actB);

    // ================= FC 8192 -> 256 (relu) ==============================
    gemm_wmma<true, _Float16><<<dim3(256 / 16, TB / 32), 32, 0, stream>>>(
        actB, 8192, fc_w, fc_b, feat, 256, 8192);

    // ================= GI = feat @ Wih^T + bih (all timesteps) ============
    gemm_wmma<false, float><<<dim3(384 / 16, TB / 32), 32, 0, stream>>>(
        feat, 256, gru_wih, gru_bih, GI, 384, 256);

    // ================= GRU recurrence =====================================
    gru_kernel<<<1, 1024, 0, stream>>>(GI, gru_whh, gru_bhh, done, gru_state, hs);

    // ================= Actor / critic heads ===============================
    heads_kernel<<<(TB * 4 + 255) / 256, 256, 0, stream>>>(
        hs, actor_w, actor_b, critic_w, critic_b, (float*)d_out);
}
